// MoE_FFN_14018773254408
// MI455X (gfx1250) — matrix operand order repaired and checked by reading, not compiled, not run on hardware
//
#include <hip/hip_runtime.h>
#include <math.h>

// ---------------- problem constants ----------------
static constexpr int Bb   = 4;
static constexpr int Ll   = 2048;
static constexpr int Cc   = 1024;
static constexpr int Hh   = 4096;
static constexpr int Ee   = 8;
static constexpr int NTOK = Bb * Ll;          // 8192 tokens
static constexpr int NASN = NTOK * 2;         // 16384 assignments (top-2)
static constexpr int MT   = 264;              // max 64-row m-tiles
static constexpr int MAXROWS = MT * 64;       // 16896
static constexpr int KC   = 64;               // K chunk (two 32-k WMMA steps)

// ---------------- workspace layout (bytes) ----------------
static constexpr size_t OFF_COUNTS  = 0;                                  // 8 int
static constexpr size_t OFF_CURSORS = 64;                                 // 8 int
static constexpr size_t OFF_OFFSETS = 128;                                // 9 int
static constexpr size_t OFF_SELE    = 256;                                // NASN int
static constexpr size_t OFF_SELW    = OFF_SELE  + (size_t)NASN * 4;       // NASN float
static constexpr size_t OFF_TROWS   = OFF_SELW  + (size_t)NASN * 4;       // NASN int
static constexpr size_t OFF_RTOK    = OFF_TROWS + (size_t)NASN * 4;       // MAXROWS int
static constexpr size_t OFF_RW      = OFF_RTOK  + (size_t)MAXROWS * 4;    // MAXROWS float
static constexpr size_t OFF_H       = OFF_RW    + (size_t)MAXROWS * 4;    // MAXROWS*H bf16
static constexpr size_t OFF_Y       = OFF_H     + (size_t)MAXROWS * Hh * 2; // MAXROWS*C f32
static constexpr size_t OFF_XBF     = OFF_Y     + (size_t)MAXROWS * Cc * 4; // NTOK*C bf16
static constexpr size_t OFF_W1T     = OFF_XBF   + (size_t)NTOK * Cc * 2;  // E*H*C bf16 (transposed)
static constexpr size_t OFF_W2T     = OFF_W1T   + (size_t)Ee * Cc * Hh * 2; // E*C*H bf16 (transposed)
static constexpr size_t WS_NEEDED   = OFF_W2T   + (size_t)Ee * Hh * Cc * 2;

// ---------------- WMMA types ----------------
typedef __bf16 v16bf __attribute__((ext_vector_type(16)));
typedef float  v8f   __attribute__((ext_vector_type(8)));

union FragU { v16bf v; uint4 q[2]; };

__device__ __forceinline__ unsigned short f2bf(float f) {
    union { float f; unsigned u; } c; c.f = f;
    unsigned u = c.u;
    unsigned r = u + 0x7FFFu + ((u >> 16) & 1u);   // round-to-nearest-even
    return (unsigned short)(r >> 16);
}

__device__ __forceinline__ unsigned long long pack4bf(float a, float b, float c, float d) {
    return (unsigned long long)f2bf(a)
         | ((unsigned long long)f2bf(b) << 16)
         | ((unsigned long long)f2bf(c) << 32)
         | ((unsigned long long)f2bf(d) << 48);
}

// A fragment: 16x32 bf16 slice at k-offset kc, LDS row-major [64][KC]
__device__ __forceinline__ v16bf load_frag_a(const unsigned short* Alds, int mrow, int kc, int lane) {
    int r = mrow + (lane & 15);
    int k = kc + (lane >> 4) * 8;
    FragU f;
    f.q[0] = *(const uint4*)(Alds + r * KC + k);
    f.q[1] = *(const uint4*)(Alds + r * KC + 16 + k);
    return f.v;
}

// B fragment: 32x16 bf16 (K x N) slice at k-offset kc, LDS N-major [256][KC]
__device__ __forceinline__ v16bf load_frag_b(const unsigned short* Blds, int ncol, int kc, int lane) {
    int cidx = ncol + (lane & 15);
    int k = kc + (lane >> 4) * 8;
    FragU f;
    f.q[0] = *(const uint4*)(Blds + cidx * KC + k);
    f.q[1] = *(const uint4*)(Blds + cidx * KC + k + 16);
    return f.v;
}

// ---------------- kernel: init ----------------
__global__ __launch_bounds__(256) void moe_init_kernel(int* counts, int* cursors, int* rowToken) {
    int idx = blockIdx.x * 256 + threadIdx.x;
    if (idx < MAXROWS) rowToken[idx] = -1;
    if (idx < Ee) { counts[idx] = 0; cursors[idx] = 0; }
}

// ---------------- kernel: x fp32 -> bf16 ----------------
__global__ __launch_bounds__(256) void moe_convx_kernel(const float* __restrict__ x,
                                                        unsigned short* __restrict__ xbf) {
    int idx = blockIdx.x * 256 + threadIdx.x;       // NTOK*C/4 threads
    float4 v = *(const float4*)(x + (size_t)idx * 4);
    *(unsigned long long*)(xbf + (size_t)idx * 4) = pack4bf(v.x, v.y, v.z, v.w);
}

// ---------------- kernel: per-expert transpose + fp32->bf16 ----------------
// src: [E][K][N] fp32   ->   dst: [E][N][K] bf16
__global__ __launch_bounds__(256) void moe_transpose_kernel(const float* __restrict__ src,
                                                            unsigned short* __restrict__ dst,
                                                            int K, int N) {
    __shared__ float tile[32][33];
    int e = blockIdx.z;
    const float* s = src + (size_t)e * K * N;
    unsigned short* d = dst + (size_t)e * K * N;
    int n0 = blockIdx.x * 32;
    int k0 = blockIdx.y * 32;
    int t = threadIdx.x;
    int r = t >> 3;            // 0..31
    int c4 = (t & 7) * 4;      // 0..28
    float4 v = *(const float4*)(s + (size_t)(k0 + r) * N + n0 + c4);
    tile[r][c4 + 0] = v.x; tile[r][c4 + 1] = v.y;
    tile[r][c4 + 2] = v.z; tile[r][c4 + 3] = v.w;
    __syncthreads();
    unsigned long long pk = pack4bf(tile[c4 + 0][r], tile[c4 + 1][r],
                                    tile[c4 + 2][r], tile[c4 + 3][r]);
    *(unsigned long long*)(d + (size_t)(n0 + r) * K + k0 + c4) = pk;
}

// ---------------- kernel: router (one wave per token) ----------------
__global__ __launch_bounds__(256) void moe_router_kernel(const float* __restrict__ x,
                                                         const float* __restrict__ gw,
                                                         const float* __restrict__ gb,
                                                         float* __restrict__ logits_out,
                                                         int* __restrict__ selE,
                                                         float* __restrict__ selW,
                                                         int* __restrict__ counts) {
    int wid  = threadIdx.x >> 5;
    int lane = threadIdx.x & 31;
    int t = blockIdx.x * 8 + wid;
    const float* xr = x + (size_t)t * Cc;
    float acc[Ee];
    #pragma unroll
    for (int e = 0; e < Ee; e++) acc[e] = 0.0f;
    for (int c = lane; c < Cc; c += 32) {
        float xv = xr[c];
        const float* g = gw + c * Ee;
        #pragma unroll
        for (int e = 0; e < Ee; e++) acc[e] = fmaf(xv, g[e], acc[e]);
    }
    #pragma unroll
    for (int e = 0; e < Ee; e++) {
        float v = acc[e];
        #pragma unroll
        for (int off = 16; off > 0; off >>= 1) v += __shfl_xor(v, off, 32);
        acc[e] = v + gb[e];
    }
    if (lane == 0) {
        float mx = acc[0];
        #pragma unroll
        for (int e = 1; e < Ee; e++) mx = fmaxf(mx, acc[e]);
        float p[Ee];
        #pragma unroll
        for (int e = 0; e < Ee; e++) p[e] = __expf(acc[e] - mx);
        int e0 = 0;
        #pragma unroll
        for (int e = 1; e < Ee; e++) if (p[e] > p[e0]) e0 = e;
        int e1 = (e0 == 0) ? 1 : 0;
        #pragma unroll
        for (int e = 0; e < Ee; e++) if (e != e0 && p[e] > p[e1]) e1 = e;
        float w0 = p[e0], w1 = p[e1];
        float inv = 1.0f / (w0 + w1);
        #pragma unroll
        for (int e = 0; e < Ee; e++) logits_out[(size_t)t * Ee + e] = acc[e];
        selE[t * 2]     = e0;
        selE[t * 2 + 1] = e1;
        selW[t * 2]     = w0 * inv;
        selW[t * 2 + 1] = w1 * inv;
        atomicAdd(&counts[e0], 1);
        atomicAdd(&counts[e1], 1);
    }
}

// ---------------- kernel: scan ----------------
__global__ void moe_scan_kernel(const int* __restrict__ counts, int* __restrict__ offs) {
    if (threadIdx.x == 0 && blockIdx.x == 0) {
        int cur = 0;
        for (int e = 0; e < Ee; e++) {
            offs[e] = cur;
            cur += ((counts[e] + 63) >> 6) << 6;
        }
        offs[Ee] = cur;
    }
}

// ---------------- kernel: scatter ----------------
__global__ __launch_bounds__(256) void moe_scatter_kernel(const int* __restrict__ selE,
                                                          const float* __restrict__ selW,
                                                          const int* __restrict__ offs,
                                                          int* __restrict__ cursors,
                                                          int* __restrict__ rowToken,
                                                          float* __restrict__ rowWeight,
                                                          int* __restrict__ tokenRows) {
    int t = blockIdx.x * 256 + threadIdx.x;
    if (t >= NTOK) return;
    #pragma unroll
    for (int k = 0; k < 2; k++) {
        int e = selE[t * 2 + k];
        int pos = atomicAdd(&cursors[e], 1);
        int row = offs[e] + pos;
        rowToken[row] = t;
        rowWeight[row] = selW[t * 2 + k];
        tokenRows[t * 2 + k] = row;
    }
}

// ---------------- kernel: layer-1 GEMM + GELU -> h (bf16) ----------------
// tile: M=64 x N=256, K over C in 64-chunks (2 WMMA k-steps per chunk).
// Padding rows (token -1) read row 0 instead: their outputs are never consumed.
__global__ __launch_bounds__(256) void moe_mlp1_kernel(const unsigned short* __restrict__ xbf,
                                                       const unsigned short* __restrict__ w1t,
                                                       const float* __restrict__ b1,
                                                       const int* __restrict__ offsets,
                                                       const int* __restrict__ rowToken,
                                                       unsigned short* __restrict__ h) {
    __shared__ unsigned short Alds[64 * KC];     // 8 KB
    __shared__ unsigned short Blds[256 * KC];    // 32 KB
    int ofs[Ee + 1];
    #pragma unroll
    for (int i = 0; i <= Ee; i++) ofs[i] = offsets[i];
    int row0 = blockIdx.x * 64;
    if (row0 >= ofs[Ee]) return;
    int e = 0;
    #pragma unroll
    for (int i = 1; i < Ee; i++) if (row0 >= ofs[i]) e = i;
    int n0 = blockIdx.y * 256;

    int tid = threadIdx.x;
    int lane = tid & 31, wid = tid >> 5;
    int mrow = (wid & 3) * 16;
    int ngrp = (wid >> 2) * 128;

    v8f acc[8] = {};
    const unsigned short* w1e = w1t + (size_t)e * Hh * Cc;   // [H][C] bf16

    int ar = tid >> 2;             // 0..63 : A staging row
    int kq = (tid & 3) * 16;       // A k sub-offset (two b128 per thread)
    int tokA = rowToken[row0 + ar];
    if (tokA < 0) tokA = 0;        // safe gather; padding output unused
    const unsigned short* xrow = xbf + (size_t)tokA * Cc;
    const unsigned short* brow = w1e + (size_t)(n0 + tid) * Cc;   // one N-row per thread

    for (int kb = 0; kb < Cc; kb += KC) {
        // stage A: two b128 per thread
        *(uint4*)(Alds + ar * KC + kq)     = *(const uint4*)(xrow + kb + kq);
        *(uint4*)(Alds + ar * KC + kq + 8) = *(const uint4*)(xrow + kb + kq + 8);
        // stage B: 128B contiguous per thread (one N-row, 64 k-values)
        #pragma unroll
        for (int u = 0; u < 8; u++)
            *(uint4*)(Blds + tid * KC + u * 8) = *(const uint4*)(brow + kb + u * 8);
        __builtin_prefetch(brow + kb + KC, 0, 1);   // next chunk of this N-row
        __syncthreads();
        #pragma unroll
        for (int kc = 0; kc < KC; kc += 32) {
            v16bf a = load_frag_a(Alds, mrow, kc, lane);
            #pragma unroll
            for (int g = 0; g < 2; g++) {
                v16bf bq[4];
                #pragma unroll
                for (int j = 0; j < 4; j++)
                    bq[j] = load_frag_b(Blds, ngrp + (g * 4 + j) * 16, kc, lane);
                #pragma unroll
                for (int j = 0; j < 4; j++)
                    acc[g * 4 + j] = __builtin_amdgcn_wmma_f32_16x16x32_bf16(
                        false, a, false, bq[j], (short)0, acc[g * 4 + j], false, false);
            }
        }
        __syncthreads();
    }
    // epilogue: + b1, exact GELU, store bf16 h
    #pragma unroll
    for (int j = 0; j < 8; j++) {
        int col = n0 + ngrp + j * 16 + (lane & 15);
        float bias = b1[e * Hh + col];
        #pragma unroll
        for (int i = 0; i < 8; i++) {
            int m = (lane >> 4) * 8 + i;
            float v = acc[j][i] + bias;
            float g = 0.5f * v * (1.0f + erff(v * 0.70710678118654752f));
            h[(size_t)(row0 + mrow + m) * Hh + col] = f2bf(g);
        }
    }
}

// ---------------- kernel: layer-2 GEMM -> weighted partials (fp32) ----------------
__global__ __launch_bounds__(256) void moe_mlp2_kernel(const unsigned short* __restrict__ h,
                                                       const unsigned short* __restrict__ w2t,
                                                       const float* __restrict__ b2,
                                                       const int* __restrict__ offsets,
                                                       const float* __restrict__ rowWeight,
                                                       float* __restrict__ y) {
    __shared__ unsigned short Alds[64 * KC];
    __shared__ unsigned short Blds[256 * KC];
    int ofs[Ee + 1];
    #pragma unroll
    for (int i = 0; i <= Ee; i++) ofs[i] = offsets[i];
    int row0 = blockIdx.x * 64;
    if (row0 >= ofs[Ee]) return;
    int e = 0;
    #pragma unroll
    for (int i = 1; i < Ee; i++) if (row0 >= ofs[i]) e = i;
    int n0 = blockIdx.y * 256;

    int tid = threadIdx.x;
    int lane = tid & 31, wid = tid >> 5;
    int mrow = (wid & 3) * 16;
    int ngrp = (wid >> 2) * 128;

    v8f acc[8] = {};
    const unsigned short* w2e = w2t + (size_t)e * Cc * Hh;   // [C][H] bf16

    int ar = tid >> 2;
    int kq = (tid & 3) * 16;
    const unsigned short* hrow = h + (size_t)(row0 + ar) * Hh;
    const unsigned short* brow = w2e + (size_t)(n0 + tid) * Hh;

    for (int kb = 0; kb < Hh; kb += KC) {
        *(uint4*)(Alds + ar * KC + kq)     = *(const uint4*)(hrow + kb + kq);
        *(uint4*)(Alds + ar * KC + kq + 8) = *(const uint4*)(hrow + kb + kq + 8);
        #pragma unroll
        for (int u = 0; u < 8; u++)
            *(uint4*)(Blds + tid * KC + u * 8) = *(const uint4*)(brow + kb + u * 8);
        __builtin_prefetch(brow + kb + KC, 0, 1);
        __syncthreads();
        #pragma unroll
        for (int kc = 0; kc < KC; kc += 32) {
            v16bf a = load_frag_a(Alds, mrow, kc, lane);
            #pragma unroll
            for (int g = 0; g < 2; g++) {
                v16bf bq[4];
                #pragma unroll
                for (int j = 0; j < 4; j++)
                    bq[j] = load_frag_b(Blds, ngrp + (g * 4 + j) * 16, kc, lane);
                #pragma unroll
                for (int j = 0; j < 4; j++)
                    acc[g * 4 + j] = __builtin_amdgcn_wmma_f32_16x16x32_bf16(
                        false, a, false, bq[j], (short)0, acc[g * 4 + j], false, false);
            }
        }
        __syncthreads();
    }
    // epilogue: (acc + b2) * combine_weight -> y partial
    float wv[8];
    #pragma unroll
    for (int i = 0; i < 8; i++)
        wv[i] = rowWeight[row0 + mrow + (lane >> 4) * 8 + i];
    #pragma unroll
    for (int j = 0; j < 8; j++) {
        int col = n0 + ngrp + j * 16 + (lane & 15);
        float bias = b2[e * Cc + col];
        #pragma unroll
        for (int i = 0; i < 8; i++) {
            int gr = row0 + mrow + (lane >> 4) * 8 + i;
            y[(size_t)gr * Cc + col] = (acc[j][i] + bias) * wv[i];
        }
    }
}

// ---------------- kernel: combine the 2 expert partials per token ----------------
__global__ __launch_bounds__(256) void moe_combine_kernel(const float* __restrict__ y,
                                                          const int* __restrict__ tokenRows,
                                                          float* __restrict__ out) {
    int idx = blockIdx.x * 256 + threadIdx.x;
    int token = idx >> 8;
    int c4 = (idx & 255) * 4;
    int rA = tokenRows[token * 2];
    int rB = tokenRows[token * 2 + 1];
    float4 a = *(const float4*)(y + (size_t)rA * Cc + c4);
    float4 b = *(const float4*)(y + (size_t)rB * Cc + c4);
    float4 o;
    o.x = a.x + b.x; o.y = a.y + b.y; o.z = a.z + b.z; o.w = a.w + b.w;
    *(float4*)(out + (size_t)token * Cc + c4) = o;
}

// ---------------- host launch ----------------
extern "C" void kernel_launch(void* const* d_in, const int* in_sizes, int n_in,
                              void* d_out, int out_size, void* d_ws, size_t ws_size,
                              hipStream_t stream) {
    const float* x      = (const float*)d_in[0];
    const float* gate_w = (const float*)d_in[1];
    const float* gate_b = (const float*)d_in[2];
    const float* w1     = (const float*)d_in[3];
    const float* b1     = (const float*)d_in[4];
    const float* w2     = (const float*)d_in[5];
    const float* b2     = (const float*)d_in[6];
    float* out = (float*)d_out;
    char* ws = (char*)d_ws;
    if (ws_size < WS_NEEDED) return;

    int*   counts    = (int*)(ws + OFF_COUNTS);
    int*   cursors   = (int*)(ws + OFF_CURSORS);
    int*   offsets   = (int*)(ws + OFF_OFFSETS);
    int*   selE      = (int*)(ws + OFF_SELE);
    float* selW      = (float*)(ws + OFF_SELW);
    int*   tokenRows = (int*)(ws + OFF_TROWS);
    int*   rowToken  = (int*)(ws + OFF_RTOK);
    float* rowWeight = (float*)(ws + OFF_RW);
    unsigned short* hbuf = (unsigned short*)(ws + OFF_H);
    float* ybuf      = (float*)(ws + OFF_Y);
    unsigned short* xbf  = (unsigned short*)(ws + OFF_XBF);
    unsigned short* w1t  = (unsigned short*)(ws + OFF_W1T);
    unsigned short* w2t  = (unsigned short*)(ws + OFF_W2T);

    float* logits_out = out + (size_t)NTOK * Cc;

    moe_init_kernel<<<(MAXROWS + 255) / 256, 256, 0, stream>>>(counts, cursors, rowToken);
    moe_convx_kernel<<<(NTOK * Cc / 4) / 256, 256, 0, stream>>>(x, xbf);
    // w1 [E][C][H] -> w1t [E][H][C]
    moe_transpose_kernel<<<dim3(Hh / 32, Cc / 32, Ee), 256, 0, stream>>>(w1, w1t, Cc, Hh);
    // w2 [E][H][C] -> w2t [E][C][H]
    moe_transpose_kernel<<<dim3(Cc / 32, Hh / 32, Ee), 256, 0, stream>>>(w2, w2t, Hh, Cc);
    moe_router_kernel<<<NTOK / 8, 256, 0, stream>>>(x, gate_w, gate_b, logits_out,
                                                    selE, selW, counts);
    moe_scan_kernel<<<1, 1, 0, stream>>>(counts, offsets);
    moe_scatter_kernel<<<(NTOK + 255) / 256, 256, 0, stream>>>(selE, selW, offsets, cursors,
                                                               rowToken, rowWeight, tokenRows);
    moe_mlp1_kernel<<<dim3(MT, Hh / 256), 256, 0, stream>>>(xbf, w1t, b1, offsets, rowToken, hbuf);
    moe_mlp2_kernel<<<dim3(MT, Cc / 256), 256, 0, stream>>>(hbuf, w2t, b2, offsets, rowWeight, ybuf);
    moe_combine_kernel<<<(NTOK * Cc / 4) / 256, 256, 0, stream>>>(ybuf, tokenRows, out);
}
